// FourierBlock_46737834115702
// MI455X (gfx1250) — matrix operand order repaired and checked by reading, not compiled, not run on hardware
//
#include <hip/hip_runtime.h>
#include <math.h>

// ---------------------------------------------------------------------------
// FourierBlock on gfx1250: rfft(64 modes) -> complex channel mix -> irfft
// expressed as three f16-WMMA GEMM stages with f32 accumulation.
//   B=16, L=1024, H=8, E=64, MODES=64
// ---------------------------------------------------------------------------

typedef __attribute__((ext_vector_type(16))) _Float16 v16h;
typedef __attribute__((ext_vector_type(8)))  _Float16 v8h;
typedef __attribute__((ext_vector_type(8)))  float    v8f;

#define WSCALE      262144.0f      // 2^18, lifts w (~3.8e-6) into f16-normal range
#define INV_WSCALE  (1.0f/262144.0f)

__device__ inline v8f zero8() {
  v8f z;
#pragma unroll
  for (int i = 0; i < 8; ++i) z[i] = 0.0f;
  return z;
}

__device__ inline v16h join_frag(v8h lo, v8h hi) {
  v16h r;
#pragma unroll
  for (int i = 0; i < 8; ++i) { r[i] = lo[i]; r[i + 8] = hi[i]; }
  return r;
}

// A-fragment (16x32 f16) from a row-major f16 matrix with leading dim `ld`.
// Layout: lane<16 -> row=lane, K={0..15}; lane>=16 -> row=lane-16,
// K={16..31}. Two b128 loads per lane.
__device__ inline v16h load_a_frag(const _Float16* base, int row0, int ld, int k0) {
  int lane = threadIdx.x & 31;
  int r    = lane & 15;
  int kh   = (lane >> 4) * 16;
  const _Float16* p = base + (size_t)(row0 + r) * ld + k0 + kh;
  return join_frag(*(const v8h*)p, *(const v8h*)(p + 8));
}

// B-fragment (32x16 f16) from fragment-ordered storage: dword index
// (subtile*32 + lane)*8 + v holds packed {K=kb+2v, K=kb+2v+1} for column
// (lane&15), kb = (lane<16 ? 0 : 16). Per lane: 16 contiguous halves.
__device__ inline v16h load_b_packed(const _Float16* base, int subtile) {
  int lane = threadIdx.x & 31;
  const _Float16* p = base + (size_t)((subtile * 32 + lane) * 8) * 2;
  return join_frag(*(const v8h*)p, *(const v8h*)(p + 8));
}

__device__ inline v8f wmma16(v16h a, v16h b, v8f c) {
  return __builtin_amdgcn_wmma_f32_16x16x32_f16(
      /*neg_a=*/false, a, /*neg_b=*/false, b,
      /*c_mod=*/(short)0, c, /*reuse_a=*/false, /*reuse_b=*/false);
}

__device__ inline unsigned pack2h(float x, float y) {
  union { _Float16 h[2]; unsigned u; } u;
  u.h[0] = (_Float16)x; u.h[1] = (_Float16)y;
  return u.u;
}

// ---------------------------------------------------------------------------
// Kernel 0: DFT bases.  F[128x1024]: rows 0..63 cos(2pi m l/L), 64..127 -sin.
// G[128x1024]: irfft basis, rows m: c_m cos, rows 64+m: -c_m sin,
// c_0=1/L (imag of DC dropped -> row 64 is zero), c_m=2/L otherwise.
// ---------------------------------------------------------------------------
__global__ __launch_bounds__(256) void k_init_dft(_Float16* F, _Float16* G) {
  int id  = blockIdx.x * 256 + threadIdx.x;       // 2 * 128 * 1024 threads
  int sel = id >> 17;                             // 0 = F, 1 = G
  int r   = id & 131071;
  int mp  = r >> 10;                              // 0..127
  int l   = r & 1023;
  int mm  = mp & 63;
  float th = 6.28318530717958647692f * (float)(mm * l) * (1.0f / 1024.0f);
  float val;
  if (sel == 0) {
    val = (mp < 64) ? __cosf(th) : -__sinf(th);
    F[(size_t)mp * 1024 + l] = (_Float16)val;
  } else {
    float c = (mm == 0) ? (1.0f / 1024.0f) : (2.0f / 1024.0f);
    if (mp < 64) val = c * __cosf(th);
    else         val = (mm == 0) ? 0.0f : -c * __sinf(th);
    G[(size_t)mp * 1024 + l] = (_Float16)val;
  }
}

// ---------------------------------------------------------------------------
// Kernel 1: repack w[h][i][o][m] (f32, m innermost) into fragment-ordered
// f16 Wt[(h,m,c)][2048 dwords], scaled by 2^18.  One dword per thread.
// ---------------------------------------------------------------------------
__global__ __launch_bounds__(256) void k_repack_w(const float* __restrict__ w_real,
                                                  const float* __restrict__ w_imag,
                                                  unsigned* __restrict__ Wt) {
  int id = blockIdx.x * 256 + threadIdx.x;        // 2^21 dwords total
  int v    = id & 7;
  int lane = (id >> 3) & 31;
  int st   = (id >> 8) & 7;                       // kstep*4 + ntile
  int c    = (id >> 11) & 1;
  int m    = (id >> 12) & 63;
  int h    = (id >> 18) & 7;
  int kstep = st >> 2, ntile = st & 3;
  int i = kstep * 32 + ((lane >> 4) * 16) + 2 * v;
  int o = ntile * 16 + (lane & 15);
  const float* w = c ? w_imag : w_real;
  size_t s0 = (((size_t)(h * 64 + i) * 64 + o) * 64) + m;
  unsigned pk = pack2h(w[s0] * WSCALE, w[s0 + 64 * 64] * WSCALE);
  Wt[(size_t)((h * 64 + m) * 2 + c) * 2048 + (id & 2047)] = pk;
}

// ---------------------------------------------------------------------------
// Kernel 2: forward DFT.  Per block (b,h): X_tile[128 x 64] = F[128x1024] @
// q[b,:,h,:][1024x64].  8 waves, each owns 16 rows of F; q tile (32x64)
// converted f32->f16 and staged in LDS in fragment order each k-step.
// Output scattered into X[(h,m,c)][b*64+e] f16 for stage 3's A-layout.
// ---------------------------------------------------------------------------
__global__ __launch_bounds__(256) void k_gemm_fwd(const float* __restrict__ q,
                                                  const _Float16* __restrict__ F,
                                                  _Float16* __restrict__ X) {
  __shared__ unsigned ldsB[1024];                 // 32(K) x 64(N) f16, packed
  int b = blockIdx.x >> 3, h = blockIdx.x & 7;
  int tid = threadIdx.x, wave = tid >> 5;
  int mbase = wave * 16;
  v8f acc[4];
#pragma unroll
  for (int j = 0; j < 4; ++j) acc[j] = zero8();

  for (int k0 = 0; k0 < 1024; k0 += 32) {
    __syncthreads();
#pragma unroll
    for (int i = 0; i < 4; ++i) {
      int idx = tid + 256 * i;                    // 1024 dwords
      int kp  = idx >> 6;                         // 0..15 (K pair / 2)
      int col = idx & 63;
      const float* src = q + ((size_t)(b * 1024 + k0 + 2 * kp)) * 512 + h * 64 + col;
      unsigned pk = pack2h(src[0], src[512]);
      int lane = (col & 15) + ((kp >= 8) ? 16 : 0);
      int j2   = col >> 4;
      ldsB[(j2 * 32 + lane) * 8 + (kp & 7)] = pk;
    }
    __syncthreads();
    v16h a = load_a_frag(F, mbase, 1024, k0);
#pragma unroll
    for (int j = 0; j < 4; ++j) {
      v16h bf = load_b_packed((const _Float16*)ldsB, j);
      acc[j] = wmma16(a, bf, acc[j]);
    }
  }

  int lane = tid & 31;
  int colL = lane & 15, hiM = (lane >> 4) * 8;
#pragma unroll
  for (int j = 0; j < 4; ++j)
#pragma unroll
    for (int r = 0; r < 8; ++r) {
      int mp = mbase + r + hiM;                   // 0..127
      int c  = mp >> 6, m = mp & 63;
      int e  = j * 16 + colL;
      X[(size_t)((h * 64 + m) * 2 + c) * 1024 + b * 64 + e] = (_Float16)acc[j][r];
    }
}

// ---------------------------------------------------------------------------
// Kernel 3: complex mode mix.  Per block (h,m): [Or|Oi](16b x 64o) =
// X(16x64) @ W(64x64) complex.  4 waves, one 16-col n-tile each.
// Or = Xr@Wr - Xi@Wi (VALU subtract),
// Oi chained through C accumulation.  Y stored f16 row=(b,h,o), k=(c,m).
// ---------------------------------------------------------------------------
__global__ __launch_bounds__(128) void k_gemm_mix(const _Float16* __restrict__ X,
                                                  const _Float16* __restrict__ Wt,
                                                  _Float16* __restrict__ Y) {
  int h = blockIdx.x >> 6, m = blockIdx.x & 63;
  int j = threadIdx.x >> 5;                       // n-tile 0..3
  const _Float16* Xr = X  + (size_t)((h * 64 + m) * 2 + 0) * 1024;
  const _Float16* Xi = X  + (size_t)((h * 64 + m) * 2 + 1) * 1024;
  const _Float16* Wr = Wt + (size_t)((h * 64 + m) * 2 + 0) * 4096;
  const _Float16* Wi = Wt + (size_t)((h * 64 + m) * 2 + 1) * 4096;

  v16h xr0 = load_a_frag(Xr, 0, 64, 0),  xr1 = load_a_frag(Xr, 0, 64, 32);
  v16h xi0 = load_a_frag(Xi, 0, 64, 0),  xi1 = load_a_frag(Xi, 0, 64, 32);
  v16h wr0 = load_b_packed(Wr, j),       wr1 = load_b_packed(Wr, 4 + j);
  v16h wi0 = load_b_packed(Wi, j),       wi1 = load_b_packed(Wi, 4 + j);

  v8f t1   = wmma16(xr1, wr1, wmma16(xr0, wr0, zero8()));   // Xr@Wr
  v8f t2   = wmma16(xi1, wi1, wmma16(xi0, wi0, zero8()));   // Xi@Wi
  v8f accI = wmma16(xi1, wr1,
             wmma16(xi0, wr0,
             wmma16(xr1, wi1,
             wmma16(xr0, wi0, zero8()))));                   // Xr@Wi + Xi@Wr
  v8f accR = t1 - t2;

  int lane = threadIdx.x & 31;
  int o = j * 16 + (lane & 15);
  int bhi = (lane >> 4) * 8;
#pragma unroll
  for (int r = 0; r < 8; ++r) {
    int bb = r + bhi;
    size_t row = (size_t)bb * 512 + h * 64 + o;  // (b,h,o) flat
    Y[row * 128 + m]      = (_Float16)(accR[r] * INV_WSCALE);
    Y[row * 128 + 64 + m] = (_Float16)(accI[r] * INV_WSCALE);
  }
}

// ---------------------------------------------------------------------------
// Kernel 4: inverse DFT.  out[8192 x 1024] = Y[8192 x 128] @ G[128 x 1024].
// Block tile 128x128; G tile staged once into 32 KB LDS in fragment order;
// 8 waves, each 16 rows x 128 cols; K = 4 steps of 32.
// ---------------------------------------------------------------------------
__global__ __launch_bounds__(256) void k_gemm_inv(const _Float16* __restrict__ Y,
                                                  const _Float16* __restrict__ G,
                                                  float* __restrict__ out) {
  __shared__ unsigned ldsB[8192];                 // 128(K) x 128(N) f16 packed
  int mt = blockIdx.x >> 3, nt = blockIdx.x & 7;
  int n0 = nt * 128;
  int tid = threadIdx.x;

#pragma unroll
  for (int i = 0; i < 32; ++i) {
    int idx = tid + 256 * i;                      // 8192 dwords
    int kp  = idx >> 7;                           // 0..63
    int col = idx & 127;
    int kstep = kp >> 4, kpl = kp & 15;
    int lane  = (col & 15) + ((kpl >= 8) ? 16 : 0);
    int ntile = col >> 4;
    const _Float16* src = G + (size_t)(kp * 2) * 1024 + n0 + col;
    ldsB[(size_t)((kstep * 8 + ntile) * 32 + lane) * 8 + (kpl & 7)] =
        pack2h((float)src[0], (float)src[1024]);
  }
  __syncthreads();

  int wave = tid >> 5;
  int mbase = mt * 128 + wave * 16;
  v8f acc[8];
#pragma unroll
  for (int j = 0; j < 8; ++j) acc[j] = zero8();

#pragma unroll
  for (int ks = 0; ks < 4; ++ks) {
    v16h a = load_a_frag(Y, mbase, 128, ks * 32);
#pragma unroll
    for (int j = 0; j < 8; ++j) {
      v16h bf = load_b_packed((const _Float16*)ldsB, ks * 8 + j);
      acc[j] = wmma16(a, bf, acc[j]);
    }
  }

  int lane = tid & 31;
  int colL = lane & 15, hiM = (lane >> 4) * 8;
#pragma unroll
  for (int j = 0; j < 8; ++j)
#pragma unroll
    for (int r = 0; r < 8; ++r) {
      size_t row = (size_t)(mbase + r + hiM);
      out[row * 1024 + n0 + j * 16 + colL] = acc[j][r];
    }
}

// ---------------------------------------------------------------------------
// Workspace map (bytes):
//   F   : [0,        256K)   f16 128x1024
//   G   : [256K,     512K)   f16 128x1024
//   X   : [512K,     2.5M)   f16 (h,m,c) x (b*64+e)        = 2 MB
//   Wt  : [2.5M,     10.5M)  f16 fragment-ordered weights  = 8 MB
//   Yb  : [10.5M,    12.5M)  f16 8192 x 128                = 2 MB
// ---------------------------------------------------------------------------
extern "C" void kernel_launch(void* const* d_in, const int* in_sizes, int n_in,
                              void* d_out, int out_size, void* d_ws, size_t ws_size,
                              hipStream_t stream) {
  const float* q      = (const float*)d_in[0];
  const float* w_real = (const float*)d_in[1];
  const float* w_imag = (const float*)d_in[2];
  float* out = (float*)d_out;

  char* ws = (char*)d_ws;
  _Float16* F  = (_Float16*)(ws);
  _Float16* G  = (_Float16*)(ws + (256ull << 10));
  _Float16* X  = (_Float16*)(ws + (512ull << 10));
  _Float16* Wt = (_Float16*)(ws + (512ull << 10) + (2ull << 20));
  _Float16* Yb = (_Float16*)(ws + (512ull << 10) + (10ull << 20));

  k_init_dft <<<1024, 256, 0, stream>>>(F, G);
  k_repack_w <<<8192, 256, 0, stream>>>(w_real, w_imag, (unsigned*)Wt);
  k_gemm_fwd <<<128,  256, 0, stream>>>(q, F, X);
  k_gemm_mix <<<512,  128, 0, stream>>>(X, Wt, Yb);
  k_gemm_inv <<<512,  256, 0, stream>>>(Yb, G, out);
}
